// SimpleGearNetModel_37220186587486
// MI455X (gfx1250) — matrix operand order repaired and checked by reading, not compiled, not run on hardware
//
#include <hip/hip_runtime.h>
#include <hip/hip_bf16.h>

// ---------------------------------------------------------------------------
// GearNet forward, MI455X (gfx1250, wave32, WMMA bf16 16x16x32).
// B=8 N=1024 D=512 E=131072 L=4.
// Rank-1 trick: edge_attr @ w1_bot == dist*u + v  (u = ew@w1_bot, v = eb@w1_bot)
// => GEMM1 shrinks to x[col] @ w1_top (K=512).
// Edge kernel: 32 edges/block (2 tiles x 16), 4 waves, each wave owns 8 N-tiles
// for BOTH edge tiles -> every B fragment feeds 2 WMMAs (2x L2 reuse).
// B loads use uniform-SGPR base (stepped per K-tile) + one static 32-bit
// voffset + immediate ioffsets -> zero divergent address VALU in the hot loop,
// which removes the WMMA->VALU WAR-hazard v_nops seen in the previous build.
// ---------------------------------------------------------------------------

#define GB   8
#define GN   1024
#define GD   512
#define GE   131072
#define GL   4
#define GBN  (GB * GN)

typedef __attribute__((ext_vector_type(16))) __bf16 v16bf;
typedef __attribute__((ext_vector_type(8)))  float  v8f;

union ABfrag {
    v16bf v;
    uint4 q[2];
};

static __device__ __forceinline__ unsigned short f2bf(float f) {
    union { float f; unsigned u; } x; x.f = f;
    unsigned r = x.u + 0x7FFFu + ((x.u >> 16) & 1u);   // round-to-nearest-even
    return (unsigned short)(r >> 16);
}

// ---------------------------------------------------------------------------
// x[i] = embed_table[tokens[i / D] * D + i % D]
// ---------------------------------------------------------------------------
__global__ void embed_kernel(const int* __restrict__ tok,
                             const float* __restrict__ tab,
                             float* __restrict__ x) {
    int i = blockIdx.x * blockDim.x + threadIdx.x;
    if (i >= GBN * GD) return;
    int t = tok[i >> 9];
    x[i] = tab[(size_t)t * GD + (i & (GD - 1))];
}

// ---------------------------------------------------------------------------
// dist[e] = || coord[row[e]] - coord[col[e]] ||  (layer-invariant)
// ---------------------------------------------------------------------------
__global__ void dist_kernel(const int* __restrict__ eidx,
                            const float* __restrict__ coord,
                            float* __restrict__ dist) {
    int e = blockIdx.x * blockDim.x + threadIdx.x;
    if (e >= GE) return;
    int r = eidx[e];
    int c = eidx[GE + e];
    float dx = coord[r * 3 + 0] - coord[c * 3 + 0];
    float dy = coord[r * 3 + 1] - coord[c * 3 + 1];
    float dz = coord[r * 3 + 2] - coord[c * 3 + 2];
    dist[e] = sqrtf(dx * dx + dy * dy + dz * dz);
}

// ---------------------------------------------------------------------------
// u[l,n]   = sum_k ew[l,k]  * w1[l, 512+k, n]
// vb1[l,n] = sum_k eb[l,k]  * w1[l, 512+k, n] + b1[l,n]
// ---------------------------------------------------------------------------
__global__ void uv_kernel(const float* __restrict__ w1,
                          const float* __restrict__ ew,
                          const float* __restrict__ eb,
                          const float* __restrict__ b1,
                          float* __restrict__ u,
                          float* __restrict__ vb1) {
    int idx = blockIdx.x * blockDim.x + threadIdx.x;   // l*D + n
    if (idx >= GL * GD) return;
    int l = idx >> 9;
    int n = idx & (GD - 1);
    const float* W  = w1 + ((size_t)l * 2 * GD + GD) * GD;   // rows 512..1023
    const float* e1 = ew + (size_t)l * GD;
    const float* e2 = eb + (size_t)l * GD;
    float su = 0.0f, sv = 0.0f;
    for (int k = 0; k < GD; ++k) {
        float w = W[(size_t)k * GD + n];
        su += e1[k] * w;
        sv += e2[k] * w;
    }
    u[idx]   = su;
    vb1[idx] = sv + b1[idx];
}

// ---------------------------------------------------------------------------
// Pack fp32 weight rows [k0, k0+K) of a [L][Ksrc][N] tensor into WMMA-B bf16
// fragment layout: wb[l][kt][nt][lane][j],
//   lane = (n%16) + 16*h, k = kt*32 + h*16 + j, n = nt*16 + n%16
// Each lane's 16 bf16 (32B) are contiguous -> two global b128 loads at use.
// ---------------------------------------------------------------------------
__global__ void pack_b_kernel(const float* __restrict__ w,
                              unsigned short* __restrict__ wb,
                              int K, int N, int L, int Ksrc, int k0) {
    int idx = blockIdx.x * blockDim.x + threadIdx.x;
    int total = L * K * N;
    if (idx >= total) return;
    int j    = idx & 15;
    int lane = (idx >> 4) & 31;
    int rest = idx >> 9;
    int ntiles = N >> 4;
    int nt = rest % ntiles;   rest /= ntiles;
    int kt = rest % (K >> 5); rest /= (K >> 5);
    int l  = rest;
    int k = k0 + kt * 32 + (lane >> 4) * 16 + j;
    int n = nt * 16 + (lane & 15);
    wb[idx] = f2bf(w[((size_t)l * Ksrc + k) * N + n]);
}

// ---------------------------------------------------------------------------
// x_next = x_cur (aggregation target), xb = bf16(x_cur) (gather source)
// ---------------------------------------------------------------------------
__global__ void prep_kernel(const float* __restrict__ xcur,
                            float* __restrict__ xnext,
                            unsigned short* __restrict__ xb) {
    int i = blockIdx.x * blockDim.x + threadIdx.x;
    if (i >= GBN * GD) return;
    float v = xcur[i];
    xnext[i] = v;
    xb[i] = f2bf(v);
}

// ---------------------------------------------------------------------------
// Fused per-edge-tile message kernel: 32 edges per block (2 A-tiles of 16).
//   A-tiles (LDS, bf16): 2 x [16 x 512] = gathered x[col]
//   GEMM1: A @ w1_top -> + dist*u + vb1 -> relu -> H (aliased into A storage)
//   GEMM2: H @ w2 -> + b2 -> atomicAdd into x_next[row]
// Wave w owns N-tiles [w*8, w*8+8) for BOTH edge tiles.
// ---------------------------------------------------------------------------
__global__ __launch_bounds__(128) void edge_kernel(
    const int* __restrict__ eidx,
    const float* __restrict__ distg,
    const unsigned short* __restrict__ xb,
    const unsigned short* __restrict__ w1b,   // packed top-half, this layer
    const unsigned short* __restrict__ w2b,   // packed, this layer
    const float* __restrict__ u,              // [D]
    const float* __restrict__ vb1,            // [D]  (v + b1)
    const float* __restrict__ b2,             // [D]
    float* __restrict__ xnext) {

    __shared__ unsigned short At[2][16][520];   // 2 tiles, 16B-aligned rows
    __shared__ int   srow[32];
    __shared__ int   scol[32];
    __shared__ float sdist[32];

    const int tid  = threadIdx.x;
    const int lane = tid & 31;
    const int wv   = tid >> 5;
    const int e0   = blockIdx.x * 32;

    if (tid < 32) {
        srow[tid]  = eidx[e0 + tid];
        scol[tid]  = eidx[GE + e0 + tid];
        sdist[tid] = distg[e0 + tid];
    }
    __syncthreads();

    // ---- Fill A-tiles: thread -> (tile, row, 128-col chunk) --------------
    {
        const int t  = tid >> 6;            // 0..1
        const int m  = (tid >> 2) & 15;     // 0..15
        const int c0 = (tid & 3) * 128;     // 0..384
        const uint4* srcp = (const uint4*)(xb + (size_t)scol[t * 16 + m] * GD + c0);
        uint4* dstp = (uint4*)&At[t][m][c0];
#pragma unroll
        for (int i = 0; i < 16; ++i) dstp[i] = srcp[i];
    }
    __syncthreads();

    const int m_a = lane & 15;   // A-matrix row held by this lane
    const int sel = lane >> 4;   // which K half-chunks this lane holds
    const int nt0 = wv * 8;      // first N-tile owned by this wave

    // loop-invariant divergent 32-bit B offset (elements); the nt*512 and
    // +8-element deltas become immediate ioffsets of the global loads.
    const unsigned laneB = (unsigned)((nt0 * 32 + lane) * 16);

    v8f acc0[8], acc1[8];
#pragma unroll
    for (int nt = 0; nt < 8; ++nt)
#pragma unroll
        for (int r = 0; r < 8; ++r) { acc0[nt][r] = 0.0f; acc1[nt][r] = 0.0f; }

    // ---- GEMM1: K = 512 -> 16 K-tiles, every B fragment used twice -------
#pragma unroll 1
    for (int kt = 0; kt < 16; ++kt) {
        const int kb = kt * 32;
        ABfrag a0, a1;
        a0.q[0] = *(const uint4*)&At[0][m_a][kb + sel * 16];
        a0.q[1] = *(const uint4*)&At[0][m_a][kb + sel * 16 + 8];
        a1.q[0] = *(const uint4*)&At[1][m_a][kb + sel * 16];
        a1.q[1] = *(const uint4*)&At[1][m_a][kb + sel * 16 + 8];
        const unsigned short* bk = w1b + ((unsigned)kt << 14);   // uniform, SGPR
#pragma unroll
        for (int nt = 0; nt < 8; ++nt) {
            const unsigned short* bp = bk + laneB + (unsigned)(nt * 512);
            ABfrag b;
            b.q[0] = *(const uint4*)bp;
            b.q[1] = *(const uint4*)(bp + 8);
            acc0[nt] = __builtin_amdgcn_wmma_f32_16x16x32_bf16(
                false, a0.v, false, b.v, (short)0, acc0[nt], false, false);
            acc1[nt] = __builtin_amdgcn_wmma_f32_16x16x32_bf16(
                false, a1.v, false, b.v, (short)0, acc1[nt], false, false);
        }
    }

    // all GEMM1 reads of the A-tiles complete before H overwrites them
    __syncthreads();

    // per-row dist / row-index registers for this lane's 8 M rows per tile
    float d0[8], d1[8];
    int   r0[8], r1[8];
#pragma unroll
    for (int r = 0; r < 8; ++r) {
        const int m = r + 8 * sel;
        d0[r] = sdist[m];        d1[r] = sdist[16 + m];
        r0[r] = srow[m];         r1[r] = srow[16 + m];
    }

    // ---- Epilogue1: h = relu(acc + dist*u + vb1) -> bf16 into A storage --
#pragma unroll
    for (int nt = 0; nt < 8; ++nt) {
        const int n = (nt0 + nt) * 16 + (lane & 15);
        const float un = u[n];
        const float vn = vb1[n];
#pragma unroll
        for (int r = 0; r < 8; ++r) {
            const int m = r + 8 * sel;
            float h0 = acc0[nt][r] + d0[r] * un + vn;
            float h1 = acc1[nt][r] + d1[r] * un + vn;
            h0 = h0 > 0.0f ? h0 : 0.0f;
            h1 = h1 > 0.0f ? h1 : 0.0f;
            At[0][m][n] = f2bf(h0);
            At[1][m][n] = f2bf(h1);
            acc0[nt][r] = 0.0f;
            acc1[nt][r] = 0.0f;
        }
    }
    __syncthreads();

    // ---- GEMM2: K = 512 -> 16 K-tiles, H read from aliased A storage -----
#pragma unroll 1
    for (int kt = 0; kt < 16; ++kt) {
        const int kb = kt * 32;
        ABfrag a0, a1;
        a0.q[0] = *(const uint4*)&At[0][m_a][kb + sel * 16];
        a0.q[1] = *(const uint4*)&At[0][m_a][kb + sel * 16 + 8];
        a1.q[0] = *(const uint4*)&At[1][m_a][kb + sel * 16];
        a1.q[1] = *(const uint4*)&At[1][m_a][kb + sel * 16 + 8];
        const unsigned short* bk = w2b + ((unsigned)kt << 14);   // uniform, SGPR
#pragma unroll
        for (int nt = 0; nt < 8; ++nt) {
            const unsigned short* bp = bk + laneB + (unsigned)(nt * 512);
            ABfrag b;
            b.q[0] = *(const uint4*)bp;
            b.q[1] = *(const uint4*)(bp + 8);
            acc0[nt] = __builtin_amdgcn_wmma_f32_16x16x32_bf16(
                false, a0.v, false, b.v, (short)0, acc0[nt], false, false);
            acc1[nt] = __builtin_amdgcn_wmma_f32_16x16x32_bf16(
                false, a1.v, false, b.v, (short)0, acc1[nt], false, false);
        }
    }

    // ---- Epilogue2: msg = acc + b2, scatter-add into x_next[row] ---------
#pragma unroll
    for (int nt = 0; nt < 8; ++nt) {
        const int n = (nt0 + nt) * 16 + (lane & 15);
        const float b2v = b2[n];
#pragma unroll
        for (int r = 0; r < 8; ++r) {
            atomicAdd(&xnext[(size_t)r0[r] * GD + n], acc0[nt][r] + b2v);
            atomicAdd(&xnext[(size_t)r1[r] * GD + n], acc1[nt][r] + b2v);
        }
    }
}

// ---------------------------------------------------------------------------
// out[0 : BN*D] = x ; out[BN*D : BN*D+BN] = (tokens == 0)
// ---------------------------------------------------------------------------
__global__ void finalize_kernel(const float* __restrict__ x,
                                const int* __restrict__ tok,
                                float* __restrict__ out) {
    int i = blockIdx.x * blockDim.x + threadIdx.x;
    if (i < GBN * GD) out[i] = x[i];
    if (i < GBN)      out[GBN * GD + i] = (tok[i] == 0) ? 1.0f : 0.0f;
}

// ---------------------------------------------------------------------------
extern "C" void kernel_launch(void* const* d_in, const int* in_sizes, int n_in,
                              void* d_out, int out_size, void* d_ws, size_t ws_size,
                              hipStream_t stream) {
    (void)in_sizes; (void)n_in; (void)out_size; (void)ws_size;

    const int*   tokens = (const int*)  d_in[0];
    const float* coords = (const float*)d_in[1];
    // d_in[2] src_distance, d_in[3] src_edge_type: unused by forward
    const int*   eidx   = (const int*)  d_in[4];
    const float* table  = (const float*)d_in[5];
    const float* edge_w = (const float*)d_in[6];   // [L,1,D]
    const float* edge_b = (const float*)d_in[7];   // [L,D]
    const float* w1     = (const float*)d_in[8];   // [L,2D,D]
    const float* b1     = (const float*)d_in[9];   // [L,D]
    const float* w2     = (const float*)d_in[10];  // [L,D,D]
    const float* b2     = (const float*)d_in[11];  // [L,D]
    float* out = (float*)d_out;

    // ---- workspace carve-up (all chunks 16B-aligned) ---------------------
    char* ws = (char*)d_ws;
    float* xA = (float*)ws;                    ws += (size_t)GBN * GD * sizeof(float);
    float* xB = (float*)ws;                    ws += (size_t)GBN * GD * sizeof(float);
    unsigned short* xb  = (unsigned short*)ws; ws += (size_t)GBN * GD * 2;
    float* distbuf = (float*)ws;               ws += (size_t)GE * sizeof(float);
    unsigned short* w1b = (unsigned short*)ws; ws += (size_t)GL * GD * GD * 2;
    unsigned short* w2b = (unsigned short*)ws; ws += (size_t)GL * GD * GD * 2;
    float* ubuf  = (float*)ws;                 ws += (size_t)GL * GD * sizeof(float);
    float* vb1buf = (float*)ws;                ws += (size_t)GL * GD * sizeof(float);

    const int nXY = GBN * GD;                 // 4,194,304

    embed_kernel<<<nXY / 256, 256, 0, stream>>>(tokens, table, xA);
    dist_kernel<<<GE / 256, 256, 0, stream>>>(eidx, coords, distbuf);
    // w1 top half (rows 0..511 of the 1024-row weight), w2 full
    pack_b_kernel<<<(GL * GD * GD) / 256, 256, 0, stream>>>(w1, w1b, GD, GD, GL, 2 * GD, 0);
    pack_b_kernel<<<(GL * GD * GD) / 256, 256, 0, stream>>>(w2, w2b, GD, GD, GL, GD, 0);
    uv_kernel<<<(GL * GD) / 256, 256, 0, stream>>>(w1, edge_w, edge_b, b1, ubuf, vb1buf);

    float* xcur = xA;
    float* xnxt = xB;
    for (int l = 0; l < GL; ++l) {
        prep_kernel<<<nXY / 256, 256, 0, stream>>>(xcur, xnxt, xb);
        edge_kernel<<<GE / 32, 128, 0, stream>>>(
            eidx, distbuf, xb,
            w1b + (size_t)l * GD * GD,
            w2b + (size_t)l * GD * GD,
            ubuf   + (size_t)l * GD,
            vb1buf + (size_t)l * GD,
            b2     + (size_t)l * GD,
            xnxt);
        float* t = xcur; xcur = xnxt; xnxt = t;
    }

    finalize_kernel<<<nXY / 256, 256, 0, stream>>>(xcur, tokens, out);
}
